// SigNet_996432413588
// MI455X (gfx1250) — matrix operand order repaired and checked by reading, not compiled, not run on hardware
//
#include <hip/hip_runtime.h>
#include <hip/hip_bf16.h>

// ---------------------------------------------------------------------------
// SigNet on gfx1250:
//   k_augment : pointwise MLP (33->64 lrelu ->8) via v_wmma_f32_16x16x32_f16
//               (unconditional clamped loads + mask -> no exec-mask churn,
//                named accumulators -> no movrel indexing)
//   k_dx      : build path increments dx (B,T,42) (inp ++ time ++ aug, diff)
//   k_sig     : fused depth-3 signature recurrence + lrelu + Wl readout.
//               grid = B*42, s3 slice + Wl3 slice live in VGPRs (7+7/thread),
//               s1/s2 recurrence redundantly in LDS per WG.
//   k_final   : deterministic reduction of 42 chunk partials -> z (B,T,1)
// ---------------------------------------------------------------------------

typedef __attribute__((ext_vector_type(16))) _Float16 v16h;
typedef __attribute__((ext_vector_type(8)))  float    v8f;

#define C_IN 33
#define H1   64
#define H2   8
#define CAUG 42
#define TT   128
#define BB   8
#define C2   (CAUG*CAUG)          // 1764
#define C3   (C2*CAUG)            // 74088

__device__ __forceinline__ float lrelu(float v) { return v >= 0.f ? v : 0.01f * v; }

// ---------------------------------------------------------------------------
// One wave (32 threads) owns a 16-row tile of B*T = 1024.
// A-matrix f16 16x32 lane layout:  K = 32*kt + 16*(h/8) + 8*(lane/16) + (h%8)
// B-matrix f16 32x16 lane layout:  K = 32*kt + 16*(h/8) + 8*(lane/16) + (h%8) , N = lane%16
// C/D f32: M = r + 8*(lane/16), N = lane%16
// ---------------------------------------------------------------------------
__global__ void __launch_bounds__(32)
k_augment(const float* __restrict__ inp,
          const float* __restrict__ W1, const float* __restrict__ b1,
          const float* __restrict__ W2, const float* __restrict__ b2,
          float* __restrict__ aug)
{
    __shared__ float hbuf[16 * 64];
    const int lane = threadIdx.x;
    const int r0   = blockIdx.x * 16;
    const int ln16 = lane & 15;
    const int hi   = lane >> 4;            // 0 / 1
    const int m    = r0 + ln16;

    // ---- stage 1: h = lrelu(inp @ W1^T + b1): 16x64 tile, K=33 pad to 64 ----
    v8f acc0, acc1, acc2, acc3;
    #pragma unroll
    for (int r = 0; r < 8; ++r) { acc0[r] = 0.f; acc1[r] = 0.f; acc2[r] = 0.f; acc3[r] = 0.f; }

    #pragma unroll
    for (int kt = 0; kt < 2; ++kt) {
        v16h a;
        #pragma unroll
        for (int h = 0; h < 16; ++h) {
            int   K   = 32 * kt + 16 * (h >> 3) + 8 * hi + (h & 7);
            int   Kc  = (K < C_IN) ? K : (C_IN - 1);     // clamp: load always valid
            float msk = (K < C_IN) ? 1.f : 0.f;          // zero the pad lanes
            a[h] = (_Float16)(inp[m * C_IN + Kc] * msk);
        }
        #pragma unroll
        for (int nt = 0; nt < 4; ++nt) {
            int n = nt * 16 + ln16;
            v16h bm;
            #pragma unroll
            for (int h = 0; h < 16; ++h) {
                int   K   = 32 * kt + 16 * (h >> 3) + 8 * hi + (h & 7);
                int   Kc  = (K < C_IN) ? K : (C_IN - 1);
                float msk = (K < C_IN) ? 1.f : 0.f;
                bm[h] = (_Float16)(W1[n * C_IN + Kc] * msk);
            }
            v8f& acc = (nt == 0) ? acc0 : (nt == 1) ? acc1 : (nt == 2) ? acc2 : acc3;
            acc = __builtin_amdgcn_wmma_f32_16x16x32_f16(
                false, a, false, bm, (short)0, acc, false, false);
        }
    }
    #pragma unroll
    for (int nt = 0; nt < 4; ++nt) {
        int n = nt * 16 + ln16;
        float bias = b1[n];
        const v8f& acc = (nt == 0) ? acc0 : (nt == 1) ? acc1 : (nt == 2) ? acc2 : acc3;
        #pragma unroll
        for (int r = 0; r < 8; ++r)
            hbuf[(r + 8 * hi) * 64 + n] = lrelu(acc[r] + bias);
    }
    __syncthreads();

    // ---- stage 2: aug = h @ W2^T + b2 : 16x8 (cols 8..15 masked zero) ----
    v8f c2;
    #pragma unroll
    for (int r = 0; r < 8; ++r) c2[r] = 0.f;
    const int   nC   = ln16 & 7;                         // clamped W2 row
    const float nmsk = (ln16 < H2) ? 1.f : 0.f;
    #pragma unroll
    for (int kt = 0; kt < 2; ++kt) {
        v16h a;
        #pragma unroll
        for (int h = 0; h < 16; ++h) {
            int K = 32 * kt + 16 * (h >> 3) + 8 * hi + (h & 7);   // K < 64 always
            a[h] = (_Float16)hbuf[ln16 * 64 + K];
        }
        v16h bm;
        #pragma unroll
        for (int h = 0; h < 16; ++h) {
            int K = 32 * kt + 16 * (h >> 3) + 8 * hi + (h & 7);   // K < 64 always
            bm[h] = (_Float16)(W2[nC * H1 + K] * nmsk);
        }
        c2 = __builtin_amdgcn_wmma_f32_16x16x32_f16(
            false, a, false, bm, (short)0, c2, false, false);
    }
    if (ln16 < H2) {
        float bias = b2[ln16];
        #pragma unroll
        for (int r = 0; r < 8; ++r) {
            int row = r0 + r + 8 * hi;
            aug[row * H2 + ln16] = c2[r] + bias;   // no activation after conv2
        }
    }
}

// ---------------------------------------------------------------------------
// Path increments with zero basepoint: dx[t] = x[t] - x[t-1], dx[0] = x[0]
// ---------------------------------------------------------------------------
__global__ void k_dx(const float* __restrict__ inp,
                     const float* __restrict__ aug,
                     float* __restrict__ dx)
{
    int idx = blockIdx.x * blockDim.x + threadIdx.x;
    if (idx >= BB * TT * CAUG) return;
    int ch = idx % CAUG;
    int t  = (idx / CAUG) % TT;
    int b  = idx / (CAUG * TT);

    auto xval = [&](int tt) -> float {
        if (ch < C_IN)  return inp[(b * TT + tt) * C_IN + ch];
        if (ch == C_IN) return (float)tt * (1.0f / (TT - 1));
        return aug[(b * TT + tt) * H2 + (ch - C_IN - 1)];
    };
    float v = xval(t);
    if (t > 0) v -= xval(t - 1);
    dx[idx] = v;
}

// ---------------------------------------------------------------------------
// Fused signature recurrence + lrelu + linear readout.
//   n3[i0,j,k] = s3 + d_k*( s2[i0,j] + d_i0*d_j/6 + 0.5*s1[i0]*d_j )
// ---------------------------------------------------------------------------
__global__ void __launch_bounds__(256)
k_sig(const float* __restrict__ dx,
      const float* __restrict__ Wl, const float* __restrict__ bl,
      float* __restrict__ partial)
{
    __shared__ float s1[CAUG];
    __shared__ float s2[C2];
    __shared__ float dbuf[CAUG];
    __shared__ float red[256];

    const int tid = threadIdx.x;
    const int i0  = blockIdx.x % CAUG;
    const int b   = blockIdx.x / CAUG;

    const bool own = (tid < 252);          // 252 * 7 = 1764 = c^2 slice
    const int  f0  = tid * 7;

    __builtin_prefetch(Wl + CAUG + C2 + i0 * C2 + ((tid * 7) % C2), 0, 0);

    // loop-invariant per-thread indices (hoisted out of the 128-step loop)
    float s3r[7], wl3[7];
    short jj[7], kk[7];                    // s3-slice (j,k) per owned element
    short uj[7], uk[7];                    // s2-update (j,k) per strided element
    #pragma unroll
    for (int u = 0; u < 7; ++u) {
        s3r[u] = 0.f; wl3[u] = 0.f;
        int f = f0 + u;
        jj[u] = (short)(f / CAUG); kk[u] = (short)(f % CAUG);
        int g = tid + 256 * u;
        uj[u] = (short)(g / CAUG); uk[u] = (short)(g % CAUG);
    }
    if (own)
        #pragma unroll
        for (int u = 0; u < 7; ++u)
            wl3[u] = Wl[CAUG + C2 + i0 * C2 + f0 + u];

    if (tid < CAUG) s1[tid] = 0.f;
    for (int f = tid; f < C2; f += 256) s2[f] = 0.f;
    __syncthreads();

    const float blv = bl[0];

    for (int t = 0; t < TT; ++t) {
        if (tid < CAUG) dbuf[tid] = dx[((b * TT) + t) * CAUG + tid];
        __syncthreads();

        const float di0  = dbuf[i0];
        const float s1i0 = s1[i0];
        float acc = 0.f;

        if (own) {
            #pragma unroll
            for (int u = 0; u < 7; ++u) {
                float dj = dbuf[jj[u]], dk = dbuf[kk[u]];
                float a  = s2[i0 * CAUG + jj[u]] + di0 * dj * (1.f / 6.f);
                float v  = fmaf(dk, fmaf(0.5f * s1i0, dj, a), s3r[u]);
                s3r[u] = v;
                acc += wl3[u] * lrelu(v);
            }
        }
        __syncthreads();                           // old s1/s2 reads complete

        #pragma unroll
        for (int u = 0; u < 7; ++u) {              // n2 = s2 + s1 (x) d + e2
            int g = tid + 256 * u;
            if (g < C2)
                s2[g] += s1[uj[u]] * dbuf[uk[u]] + 0.5f * dbuf[uj[u]] * dbuf[uk[u]];
        }
        __syncthreads();
        if (tid < CAUG) s1[tid] += dbuf[tid];      // n1 = s1 + d
        __syncthreads();

        if (i0 == 0) {                             // level-1/2 readout + bias
            #pragma unroll
            for (int u = 0; u < 7; ++u) {
                int g = tid + 256 * u;
                if (g < C2) acc += Wl[CAUG + g] * lrelu(s2[g]);
            }
            if (tid < CAUG) acc += Wl[tid] * lrelu(s1[tid]);
            if (tid == 0)   acc += blv;
        }

        red[tid] = acc;                            // deterministic tree reduce
        __syncthreads();
        for (int s = 128; s > 0; s >>= 1) {
            if (tid < s) red[tid] += red[tid + s];
            __syncthreads();
        }
        if (tid == 0) partial[(b * CAUG + i0) * TT + t] = red[0];
        __syncthreads();
    }
}

// ---------------------------------------------------------------------------
__global__ void k_final(const float* __restrict__ partial, float* __restrict__ out)
{
    int idx = blockIdx.x * blockDim.x + threadIdx.x;
    if (idx >= BB * TT) return;
    int b = idx / TT, t = idx % TT;
    float s = 0.f;
    for (int c = 0; c < CAUG; ++c) s += partial[(b * CAUG + c) * TT + t];
    out[idx] = s;
}

// ---------------------------------------------------------------------------
extern "C" void kernel_launch(void* const* d_in, const int* in_sizes, int n_in,
                              void* d_out, int out_size, void* d_ws, size_t ws_size,
                              hipStream_t stream)
{
    const float* inp = (const float*)d_in[0];   // (8,128,33)
    const float* W1  = (const float*)d_in[1];   // (64,33)
    const float* b1  = (const float*)d_in[2];   // (64)
    const float* W2  = (const float*)d_in[3];   // (8,64)
    const float* b2  = (const float*)d_in[4];   // (8)
    const float* Wl  = (const float*)d_in[5];   // (1,75894)
    const float* bl  = (const float*)d_in[6];   // (1)
    float* out = (float*)d_out;                 // (8,128,1) = 1024

    float* ws      = (float*)d_ws;
    float* aug     = ws;                        //  8192 floats
    float* dxb     = ws + 8192;                 // 43008 floats
    float* partial = ws + 8192 + 43008;         // 43008 floats  (~369 KB total)

    k_augment<<<(BB * TT) / 16, 32, 0, stream>>>(inp, W1, b1, W2, b2, aug);
    k_dx<<<(BB * TT * CAUG + 255) / 256, 256, 0, stream>>>(inp, aug, dxb);
    k_sig<<<BB * CAUG, 256, 0, stream>>>(dxb, Wl, bl, partial);
    k_final<<<(BB * TT + 255) / 256, 256, 0, stream>>>(partial, out);
}
